// DecoderForClassification_36335423324713
// MI455X (gfx1250) — matrix operand order repaired and checked by reading, not compiled, not run on hardware
//
#include <hip/hip_runtime.h>

// ---------------- problem constants ----------------
constexpr int B_  = 4;
constexpr int S_  = 1024;
constexpr int E_  = 1024;
constexpr int H_  = 16;
constexpr int DH_ = 64;
constexpr int L_  = 8;
constexpr int FF_ = 4096;
constexpr int C_  = 10;
constexpr int M_  = B_ * S_;          // 4096 token rows

// ---------------- types ----------------
typedef __attribute__((ext_vector_type(16))) __bf16        v16bf;
typedef __attribute__((ext_vector_type(8)))  float         v8f;
typedef __attribute__((ext_vector_type(4)))  unsigned int  u32x4;
typedef __attribute__((ext_vector_type(8)))  unsigned int  u32x8;
typedef __attribute__((ext_vector_type(4)))  int           i32x4;
typedef __attribute__((ext_vector_type(8)))  int           i32x8;

__device__ __forceinline__ unsigned short f2bf(float f) {
    unsigned int u = __builtin_bit_cast(unsigned int, f);
    u += 0x7FFFu + ((u >> 16) & 1u);          // round-to-nearest-even
    return (unsigned short)(u >> 16);
}
__device__ __forceinline__ float bf2f(unsigned short h) {
    unsigned int u = ((unsigned int)h) << 16;
    return __builtin_bit_cast(float, u);
}

// ---------------- Tensor Data Mover: 2D tile global->LDS with LDS padding ----------------
// rows = tile_dim1, cols = tile_dim0 (bf16 elements), rowStride in elements.
// pad codes: interval c -> pad after 2^(c+1) DWORDs; amount a -> (a+1) DWORDs inserted.
__device__ __forceinline__ void tdm_load_2d(unsigned lds_off, const void* gaddr,
                                            unsigned rows, unsigned cols,
                                            unsigned long long rowStride,
                                            unsigned padIntervalCode, unsigned padAmountCode) {
    unsigned long long ga = (unsigned long long)(uintptr_t)gaddr;
    u32x4 g0;
    g0[0] = 1u;                                               // count=1 valid descriptor
    g0[1] = lds_off;                                          // LDS byte address
    g0[2] = (unsigned)(ga & 0xFFFFFFFFu);                     // global_addr[31:0]
    g0[3] = (unsigned)((ga >> 32) & 0x1FFFFFFu) | (2u << 30); // global_addr[56:32] | type=2
    i32x8 g1;
    unsigned w0 = (1u << 16)                                  // data_size = 2 bytes (bf16)
                | (1u << 20)                                  // pad_enable
                | (padIntervalCode << 22)
                | (padAmountCode << 25);
    g1[0] = (int)w0;
    g1[1] = (int)((cols & 0xFFFFu) << 16);                    // tensor_dim0[15:0]
    g1[2] = (int)(((cols >> 16) & 0xFFFFu) | ((rows & 0xFFFFu) << 16));   // dim0 hi | dim1 lo
    g1[3] = (int)(((rows >> 16) & 0xFFFFu) | ((cols & 0xFFFFu) << 16));   // dim1 hi | tile_dim0
    g1[4] = (int)(rows & 0xFFFFu);                            // tile_dim1 ; tile_dim2 = 0
    g1[5] = (int)(unsigned)(rowStride & 0xFFFFFFFFull);       // tensor_dim0_stride[31:0]
    g1[6] = (int)(unsigned)((rowStride >> 32) & 0xFFFFull);   // stride[47:32]
    g1[7] = 0;
    i32x4 z4 = {0, 0, 0, 0};
#if __clang_major__ >= 23
    i32x8 z8 = {0, 0, 0, 0, 0, 0, 0, 0};
    __builtin_amdgcn_tensor_load_to_lds(g0, g1, z4, z4, z8, 0);
#else
    __builtin_amdgcn_tensor_load_to_lds(g0, g1, z4, z4, 0);
#endif
}

// ---------------- WMMA fragment gathers (wave32 layouts) ----------------
__device__ __forceinline__ v16bf load_frag_a(const unsigned short* p, int ldh) {
    int lane = threadIdx.x & 31;
    int m = lane & 15, g = lane >> 4;
    const unsigned short* row = p + m * ldh;
    u32x4 lo = *(const u32x4*)(row + 8 * g);
    u32x4 hi = *(const u32x4*)(row + 16 + 8 * g);
    u32x8 v;
    v[0]=lo[0]; v[1]=lo[1]; v[2]=lo[2]; v[3]=lo[3];
    v[4]=hi[0]; v[5]=hi[1]; v[6]=hi[2]; v[7]=hi[3];
    return __builtin_bit_cast(v16bf, v);
}
__device__ __forceinline__ v16bf load_frag_b(const unsigned short* p, int ldh) {
    int lane = threadIdx.x & 31;
    int n = lane & 15, g = lane >> 4;
    const unsigned short* row = p + n * ldh + 8 * g;
    u32x4 lo = *(const u32x4*)(row);
    u32x4 hi = *(const u32x4*)(row + 16);
    u32x8 v;
    v[0]=lo[0]; v[1]=lo[1]; v[2]=lo[2]; v[3]=lo[3];
    v[4]=hi[0]; v[5]=hi[1]; v[6]=hi[2]; v[7]=hi[3];
    return __builtin_bit_cast(v16bf, v);
}
__device__ __forceinline__ v8f wmma_bf16(v16bf a, v16bf b, v8f c) {
    return __builtin_amdgcn_wmma_f32_16x16x32_bf16(false, a, false, b, (short)0, c, false, false);
}

// ---------------- weight transpose + fp32->bf16: src [z][R][C] -> dst [z][C][R] ----------------
__global__ __launch_bounds__(256) void transpose_bf16_k(const float* __restrict__ src,
                                                        unsigned short* __restrict__ dst,
                                                        int R, int Cn) {
    __shared__ float tile[32][33];
    int z = blockIdx.z;
    const float* s = src + (size_t)z * R * Cn;
    unsigned short* d = dst + (size_t)z * R * Cn;
    int c0 = blockIdx.x * 32, r0 = blockIdx.y * 32;
    int tx = threadIdx.x & 31, ty = threadIdx.x >> 5;     // 8 rows/pass, 4 passes
    for (int i = 0; i < 4; ++i)
        tile[ty + 8 * i][tx] = s[(size_t)(r0 + ty + 8 * i) * Cn + c0 + tx];
    __syncthreads();
    for (int i = 0; i < 4; ++i)
        d[(size_t)(c0 + ty + 8 * i) * R + r0 + tx] = f2bf(tile[tx][ty + 8 * i]);
}

// ---------------- embedding + positional encoding (fp32 h) ----------------
__global__ __launch_bounds__(256) void embed_pe_k(const int* __restrict__ x,
                                                  const float* __restrict__ emb,
                                                  const float* __restrict__ pe,
                                                  float* __restrict__ h) {
    int r = blockIdx.x, tid = threadIdx.x;
    int s = r & (S_ - 1);
    int tok = x[r];
    for (int i = 0; i < 4; ++i) {
        int e = tid + 256 * i;
        h[(size_t)r * E_ + e] = emb[(size_t)tok * E_ + e] * 32.0f + pe[(size_t)s * E_ + e];
    }
}

// ---------------- layernorm over E=1024, bf16 output ----------------
__global__ __launch_bounds__(256) void layernorm_k(const float* __restrict__ in,
                                                   const float* __restrict__ w,
                                                   const float* __restrict__ bb,
                                                   unsigned short* __restrict__ out) {
    int row = blockIdx.x, tid = threadIdx.x;
    __shared__ float red[256];
    __shared__ float s_mu, s_rstd;
    const float* x = in + (size_t)row * E_;
    float v[4], sum = 0.f;
    for (int i = 0; i < 4; ++i) { v[i] = x[tid + 256 * i]; sum += v[i]; }
    red[tid] = sum; __syncthreads();
    for (int off = 128; off > 0; off >>= 1) { if (tid < off) red[tid] += red[tid + off]; __syncthreads(); }
    if (tid == 0) s_mu = red[0] * (1.0f / E_);
    __syncthreads();
    float mu = s_mu, vs = 0.f;
    for (int i = 0; i < 4; ++i) { float d = v[i] - mu; vs += d * d; }
    red[tid] = vs; __syncthreads();
    for (int off = 128; off > 0; off >>= 1) { if (tid < off) red[tid] += red[tid + off]; __syncthreads(); }
    if (tid == 0) s_rstd = rsqrtf(red[0] * (1.0f / E_) + 1e-5f);
    __syncthreads();
    float rs = s_rstd;
    for (int i = 0; i < 4; ++i) {
        int e = tid + 256 * i;
        out[(size_t)row * E_ + e] = f2bf((v[i] - mu) * rs * w[e] + bb[e]);
    }
}

// ---------------- TDM-staged, double-buffered bf16 WMMA GEMM ----------------
// Out = act( A(MxK bf16) * Bt[z]^T + bias + Res );  Bt is [N][K] bf16 (pre-transposed).
constexpr int BM = 128, BN = 64, BKS = 64, LPA = 72;   // LDS row stride 72 halves (144B, 16B aligned)

__global__ __launch_bounds__(256) void gemm_bf16_tdm(
    const unsigned short* __restrict__ A, int lda,
    const unsigned short* __restrict__ Bt, long long bz, int ldbt,
    float* __restrict__ OutF, long long ozF, int ldoF,
    unsigned short* __restrict__ OutB, long long ozB, int ldoB, int outT,
    const float* __restrict__ bias, long long biasz,
    const float* __restrict__ Res, int ldr,
    int K, int relu) {
    __shared__ alignas(16) unsigned short Al[2][BM][LPA];
    __shared__ alignas(16) unsigned short Bl[2][BN][LPA];
    int tid = threadIdx.x, lane = tid & 31, wv = tid >> 5;
    int wm = wv >> 1, wn = wv & 1;                       // 4 x 2 wave grid -> 32x32/wave
    int z = blockIdx.z;
    const unsigned short* Bz = Bt + (size_t)z * bz;
    int m0 = blockIdx.y * BM, n0 = blockIdx.x * BN;
    int nT = K / BKS;

    v8f zero = {0.f,0.f,0.f,0.f,0.f,0.f,0.f,0.f};
    v8f acc[2][2];
    for (int i = 0; i < 2; ++i) for (int j = 0; j < 2; ++j) acc[i][j] = zero;

    unsigned aOff0 = (unsigned)(uintptr_t)&Al[0][0][0];
    unsigned aOff1 = (unsigned)(uintptr_t)&Al[1][0][0];
    unsigned bOff0 = (unsigned)(uintptr_t)&Bl[0][0][0];
    unsigned bOff1 = (unsigned)(uintptr_t)&Bl[1][0][0];

    if (wv == 0) {   // prologue: stage tile 0 into buffer 0 (wave-level TDM issue)
        tdm_load_2d(aOff0, A + (size_t)m0 * lda, BM, BKS, (unsigned long long)lda, 4, 3);
        tdm_load_2d(bOff0, Bz + (size_t)n0 * ldbt, BN, BKS, (unsigned long long)ldbt, 4, 3);
    }
    for (int t = 0; t < nT; ++t) {
        if (wv == 0) {
            if (t + 1 < nT) {
                unsigned ao = ((t + 1) & 1) ? aOff1 : aOff0;
                unsigned bo = ((t + 1) & 1) ? bOff1 : bOff0;
                size_t k1 = (size_t)(t + 1) * BKS;
                tdm_load_2d(ao, A + (size_t)m0 * lda + k1, BM, BKS, (unsigned long long)lda, 4, 3);
                tdm_load_2d(bo, Bz + (size_t)n0 * ldbt + k1, BN, BKS, (unsigned long long)ldbt, 4, 3);
                __builtin_amdgcn_s_wait_tensorcnt(2);    // tile t landed; t+1 in flight
            } else {
                __builtin_amdgcn_s_wait_tensorcnt(0);
            }
        }
        __syncthreads();
        int bf = t & 1;
        for (int kc = 0; kc < 2; ++kc) {                 // two 32-deep WMMA steps per stage
            v16bf a0 = load_frag_a(&Al[bf][wm * 32 +  0][32 * kc], LPA);
            v16bf a1 = load_frag_a(&Al[bf][wm * 32 + 16][32 * kc], LPA);
            v16bf b0 = load_frag_b(&Bl[bf][wn * 32 +  0][32 * kc], LPA);
            v16bf b1 = load_frag_b(&Bl[bf][wn * 32 + 16][32 * kc], LPA);
            acc[0][0] = wmma_bf16(a0, b0, acc[0][0]);
            acc[0][1] = wmma_bf16(a0, b1, acc[0][1]);
            acc[1][0] = wmma_bf16(a1, b0, acc[1][0]);
            acc[1][1] = wmma_bf16(a1, b1, acc[1][1]);
        }
        __syncthreads();
    }

    const float* Bi = bias ? bias + (size_t)z * biasz : nullptr;
    int cn = lane & 15, g = lane >> 4;
    for (int i = 0; i < 2; ++i)
        for (int j = 0; j < 2; ++j)
            for (int r = 0; r < 8; ++r) {
                int row = m0 + wm * 32 + 16 * i + r + 8 * g;
                int col = n0 + wn * 32 + 16 * j + cn;
                float v = acc[i][j][r];
                if (Bi)   v += Bi[col];
                if (Res)  v += Res[(size_t)row * ldr + col];
                if (relu) v = fmaxf(v, 0.f);
                if (OutF) OutF[(size_t)z * ozF + (size_t)row * ldoF + col] = v;
                if (OutB) {
                    unsigned short bv = f2bf(v);
                    if (outT) OutB[(size_t)z * ozB + (size_t)col * ldoB + row] = bv;
                    else      OutB[(size_t)z * ozB + (size_t)row * ldoB + col] = bv;
                }
            }
}

// ---------------- flash attention, TDM-staged tiles (one WG = 64 q rows of one (b,h)) ----------------
// Qb/Kb: [H][M][DH] bf16 ; VTb: [H][DH][M] bf16 (V transposed) ; out: [M][E] bf16
__global__ __launch_bounds__(128) void attention_wmma(
    const unsigned short* __restrict__ Qb, const unsigned short* __restrict__ Kb,
    const unsigned short* __restrict__ VTb,
    const unsigned char* __restrict__ mask, unsigned short* __restrict__ O) {
    __shared__ alignas(16) unsigned short ql[64][72];
    __shared__ alignas(16) unsigned short kl[32][72];
    __shared__ alignas(16) unsigned short vl[64][40];   // [d][k]
    __shared__ alignas(16) unsigned short pl[64][40];   // probs bf16
    __shared__ float sS[64][33];
    __shared__ float rowM[64], rowL[64], rowScale[64], pm[64][2], ps[64][2];

    int tid = threadIdx.x, lane = tid & 31, wv = tid >> 5;
    int q0 = blockIdx.x * 64, h = blockIdx.y, b = blockIdx.z;
    size_t base = (size_t)h * M_ + (size_t)b * S_;

    if (wv == 0) {      // Q tile 64x64 via TDM
        tdm_load_2d((unsigned)(uintptr_t)&ql[0][0], Qb + (base + q0) * DH_,
                    64, 64, (unsigned long long)DH_, 4, 3);
        __builtin_amdgcn_s_wait_tensorcnt(0);
    }
    if (tid < 64) { rowM[tid] = -1e30f; rowL[tid] = 0.f; }
    __syncthreads();

    v16bf aQ0 = load_frag_a(&ql[wv * 16][0],  72);
    v16bf aQ1 = load_frag_a(&ql[wv * 16][32], 72);
    v8f zero = {0.f,0.f,0.f,0.f,0.f,0.f,0.f,0.f};
    v8f accO[4];
    for (int j = 0; j < 4; ++j) accO[j] = zero;

    for (int kblk = 0; kblk < S_ / 32; ++kblk) {
        __syncthreads();                                // previous tile fully consumed
        if (wv == 0) {
            tdm_load_2d((unsigned)(uintptr_t)&kl[0][0], Kb + (base + kblk * 32) * DH_,
                        32, 64, (unsigned long long)DH_, 4, 3);
            tdm_load_2d((unsigned)(uintptr_t)&vl[0][0],
                        VTb + (size_t)h * DH_ * M_ + (size_t)b * S_ + kblk * 32,
                        64, 32, (unsigned long long)M_, 3, 3);
            __builtin_amdgcn_s_wait_tensorcnt(0);
        }
        __syncthreads();
        // scores = (Q K^T) * 1/sqrt(DH)
        {
            int n = lane & 15, g = lane >> 4;
            for (int t = 0; t < 2; ++t) {
                v16bf bK0 = load_frag_b(&kl[16 * t][0],  72);
                v16bf bK1 = load_frag_b(&kl[16 * t][32], 72);
                v8f s = zero;
                s = wmma_bf16(aQ0, bK0, s);
                s = wmma_bf16(aQ1, bK1, s);
                for (int r = 0; r < 8; ++r)
                    sS[wv * 16 + r + 8 * g][16 * t + n] = s[r] * 0.125f;
            }
        }
        __syncthreads();
        // online softmax: 2 threads/row, 16 cols each
        int row = tid >> 1, half = tid & 1;
        float vals[16];
        {
            float mx = -1e30f;
            for (int i = 0; i < 16; ++i) {
                int c = half * 16 + i;
                int kpos = kblk * 32 + c;
                float v = mask[(size_t)b * S_ + kpos] ? -1e30f : sS[row][c];
                vals[i] = v;
                mx = fmaxf(mx, v);
            }
            pm[row][half] = mx;
        }
        __syncthreads();
        if (half == 0) {
            float nm = fmaxf(rowM[row], fmaxf(pm[row][0], pm[row][1]));
            rowScale[row] = __expf(rowM[row] - nm);
            rowM[row] = nm;
        }
        __syncthreads();
        {
            float sum = 0.f, nm = rowM[row];
            for (int i = 0; i < 16; ++i) {
                int c = half * 16 + i;
                float p = __expf(vals[i] - nm);
                pl[row][c] = f2bf(p);
                sum += p;
            }
            ps[row][half] = sum;
        }
        __syncthreads();
        if (half == 0) rowL[row] = rowL[row] * rowScale[row] + ps[row][0] + ps[row][1];
        __syncthreads();
        // rescale running output, accumulate P * V
        {
            int g = lane >> 4;
            float sc[8];
            for (int r = 0; r < 8; ++r) sc[r] = rowScale[wv * 16 + r + 8 * g];
            v16bf aP = load_frag_a(&pl[wv * 16][0], 40);
            for (int j = 0; j < 4; ++j) {
                for (int r = 0; r < 8; ++r) accO[j][r] *= sc[r];
                v16bf bV = load_frag_b(&vl[16 * j][0], 40);
                accO[j] = wmma_bf16(aP, bV, accO[j]);
            }
        }
    }
    __syncthreads();
    {
        int n = lane & 15, g = lane >> 4;
        for (int j = 0; j < 4; ++j)
            for (int r = 0; r < 8; ++r) {
                int lr = wv * 16 + r + 8 * g;
                int srow = q0 + lr;
                O[((size_t)b * S_ + srow) * E_ + h * DH_ + 16 * j + n] =
                    f2bf(accO[j][r] / rowL[lr]);
            }
    }
}

// ---------------- masked last-token gather + classifier (bf16 in, fp32 out) ----------------
__global__ __launch_bounds__(256) void classify_k(const unsigned short* __restrict__ xn,
                                                  const unsigned char* __restrict__ mask,
                                                  const float* __restrict__ cw,
                                                  const float* __restrict__ cb,
                                                  float* __restrict__ out) {
    int b = blockIdx.x, tid = threadIdx.x;
    __shared__ float red[256];
    __shared__ int s_idx;
    int cnt = 0;
    for (int s = tid; s < S_; s += 256) cnt += (mask[(size_t)b * S_ + s] == 0);
    red[tid] = (float)cnt; __syncthreads();
    for (int off = 128; off > 0; off >>= 1) { if (tid < off) red[tid] += red[tid + off]; __syncthreads(); }
    if (tid == 0) { int len = (int)red[0]; s_idx = len > 0 ? len - 1 : 0; }
    __syncthreads();
    const unsigned short* rowp = xn + ((size_t)b * S_ + s_idx) * E_;
    float loc[C_];
    for (int c = 0; c < C_; ++c) loc[c] = 0.f;
    for (int e = tid; e < E_; e += 256) {
        float xe = bf2f(rowp[e]);
        for (int c = 0; c < C_; ++c) loc[c] += xe * cw[(size_t)e * C_ + c];
    }
    for (int c = 0; c < C_; ++c) {
        red[tid] = loc[c]; __syncthreads();
        for (int off = 128; off > 0; off >>= 1) { if (tid < off) red[tid] += red[tid + off]; __syncthreads(); }
        if (tid == 0) out[b * C_ + c] = red[0] + cb[c];
        __syncthreads();
    }
}

// ---------------- host driver ----------------
extern "C" void kernel_launch(void* const* d_in, const int* in_sizes, int n_in,
                              void* d_out, int out_size, void* d_ws, size_t ws_size,
                              hipStream_t stream) {
    (void)in_sizes; (void)n_in; (void)out_size; (void)ws_size;
    const int*           x    = (const int*)d_in[0];
    const unsigned char* mask = (const unsigned char*)d_in[1];
    const float* emb  = (const float*)d_in[2];
    const float* pe   = (const float*)d_in[3];
    const float* WQ   = (const float*)d_in[4];
    const float* bQ   = (const float*)d_in[5];
    const float* WK   = (const float*)d_in[6];
    const float* bK   = (const float*)d_in[7];
    const float* WV   = (const float*)d_in[8];
    const float* bV   = (const float*)d_in[9];
    const float* WO   = (const float*)d_in[10];
    const float* ln1w = (const float*)d_in[11];
    const float* ln1b = (const float*)d_in[12];
    const float* ln2w = (const float*)d_in[13];
    const float* ln2b = (const float*)d_in[14];
    const float* f1w  = (const float*)d_in[15];
    const float* f1b  = (const float*)d_in[16];
    const float* f2w  = (const float*)d_in[17];
    const float* f2b  = (const float*)d_in[18];
    const float* flnw = (const float*)d_in[19];
    const float* flnb = (const float*)d_in[20];
    const float* clsw = (const float*)d_in[21];
    const float* clsb = (const float*)d_in[22];

    // ---- workspace carve (byte-based, 256B aligned chunks) ----
    char* p = (char*)d_ws;
    auto carve = [&](size_t bytes) { char* r = p; p += (bytes + 255) & ~(size_t)255; return r; };
    float*          h    = (float*)         carve((size_t)M_ * E_ * 4);       // fp32 residual stream
    unsigned short* xnb  = (unsigned short*)carve((size_t)M_ * E_ * 2);       // bf16 activations
    unsigned short* Qb   = (unsigned short*)carve((size_t)H_ * M_ * DH_ * 2);
    unsigned short* Kb   = (unsigned short*)carve((size_t)H_ * M_ * DH_ * 2);
    unsigned short* VTb  = (unsigned short*)carve((size_t)H_ * DH_ * M_ * 2); // V transposed
    unsigned short* mid  = (unsigned short*)carve((size_t)M_ * FF_ * 2);
    unsigned short* WQt  = (unsigned short*)carve((size_t)L_ * H_ * E_ * DH_ * 2);
    unsigned short* WKt  = (unsigned short*)carve((size_t)L_ * H_ * E_ * DH_ * 2);
    unsigned short* WVt  = (unsigned short*)carve((size_t)L_ * H_ * E_ * DH_ * 2);
    unsigned short* WOt  = (unsigned short*)carve((size_t)L_ * E_ * E_ * 2);
    unsigned short* f1t  = (unsigned short*)carve((size_t)L_ * E_ * FF_ * 2);
    unsigned short* f2t  = (unsigned short*)carve((size_t)L_ * FF_ * E_ * 2);

    // ---- one-time weight transpose+convert: [z][R][C] fp32 -> [z][C][R] bf16 ----
    transpose_bf16_k<<<dim3(DH_/32, E_/32, L_*H_), 256, 0, stream>>>(WQ,  WQt, E_,  DH_);
    transpose_bf16_k<<<dim3(DH_/32, E_/32, L_*H_), 256, 0, stream>>>(WK,  WKt, E_,  DH_);
    transpose_bf16_k<<<dim3(DH_/32, E_/32, L_*H_), 256, 0, stream>>>(WV,  WVt, E_,  DH_);
    transpose_bf16_k<<<dim3(E_/32,  E_/32, L_),    256, 0, stream>>>(WO,  WOt, E_,  E_);
    transpose_bf16_k<<<dim3(FF_/32, E_/32, L_),    256, 0, stream>>>(f1w, f1t, E_,  FF_);
    transpose_bf16_k<<<dim3(E_/32,  FF_/32, L_),   256, 0, stream>>>(f2w, f2t, FF_, E_);

    embed_pe_k<<<M_, 256, 0, stream>>>(x, emb, pe, h);

    const long long wQKVz = (long long)E_ * DH_;    // per-head transposed weight stride
    for (int l = 0; l < L_; ++l) {
        layernorm_k<<<M_, 256, 0, stream>>>(h, ln1w + (size_t)l * E_, ln1b + (size_t)l * E_, xnb);

        dim3 gq(DH_ / BN, M_ / BM, H_);
        gemm_bf16_tdm<<<gq, 256, 0, stream>>>(xnb, E_,
            WQt + (size_t)l * H_ * E_ * DH_, wQKVz, E_,
            nullptr, 0, 0, Qb, (long long)M_ * DH_, DH_, 0,
            bQ + (size_t)l * H_ * DH_, DH_, nullptr, 0, E_, 0);
        gemm_bf16_tdm<<<gq, 256, 0, stream>>>(xnb, E_,
            WKt + (size_t)l * H_ * E_ * DH_, wQKVz, E_,
            nullptr, 0, 0, Kb, (long long)M_ * DH_, DH_, 0,
            bK + (size_t)l * H_ * DH_, DH_, nullptr, 0, E_, 0);
        gemm_bf16_tdm<<<gq, 256, 0, stream>>>(xnb, E_,        // V written transposed [DH][M]
            WVt + (size_t)l * H_ * E_ * DH_, wQKVz, E_,
            nullptr, 0, 0, VTb, (long long)DH_ * M_, M_, 1,
            bV + (size_t)l * H_ * DH_, DH_, nullptr, 0, E_, 0);

        dim3 ga(S_ / 64, H_, B_);
        attention_wmma<<<ga, 128, 0, stream>>>(Qb, Kb, VTb, mask, xnb);

        dim3 go(E_ / BN, M_ / BM, 1);
        gemm_bf16_tdm<<<go, 256, 0, stream>>>(xnb, E_,
            WOt + (size_t)l * E_ * E_, 0, E_,
            h, 0, E_, nullptr, 0, 0, 0,
            nullptr, 0, h, E_, E_, 0);

        layernorm_k<<<M_, 256, 0, stream>>>(h, ln2w + (size_t)l * E_, ln2b + (size_t)l * E_, xnb);

        dim3 g1(FF_ / BN, M_ / BM, 1);
        gemm_bf16_tdm<<<g1, 256, 0, stream>>>(xnb, E_,
            f1t + (size_t)l * E_ * FF_, 0, E_,
            nullptr, 0, 0, mid, 0, FF_, 0,
            f1b + (size_t)l * FF_, 0, nullptr, 0, E_, 1);

        dim3 g2(E_ / BN, M_ / BM, 1);
        gemm_bf16_tdm<<<g2, 256, 0, stream>>>(mid, FF_,
            f2t + (size_t)l * FF_ * E_, 0, FF_,
            h, 0, E_, nullptr, 0, 0, 0,
            f2b + (size_t)l * E_, 0, h, E_, FF_, 0);
    }

    layernorm_k<<<M_, 256, 0, stream>>>(h, flnw, flnb, xnb);
    classify_k<<<B_, 256, 0, stream>>>(xnb, mask, clsw, clsb, (float*)d_out);
}
